// GetModel_4011499454799
// MI455X (gfx1250) — matrix operand order repaired and checked by reading, not compiled, not run on hardware
//
#include <hip/hip_runtime.h>
#include <hip/hip_bf16.h>
#include <stdint.h>

typedef __attribute__((ext_vector_type(16))) _Float16 v16h;
typedef __attribute__((ext_vector_type(8)))  _Float16 v8h;
typedef __attribute__((ext_vector_type(8)))  float    v8f;

#define BN_EPS 1e-5f

// --------------------------------------------------------------------------
// Weight pre-pack: W[K,N] f32 (row-major)  ->  Wp[kb][n][k32] f16,
// K padded to Kpad (mult of 32), N padded to Npad (mult of 64), zero-filled.
// A lane's B-fragment (k = h*16 .. h*16+15, fixed n) is then 2 contiguous,
// 32B-aligned b128 loads.
// --------------------------------------------------------------------------
__global__ void pack_w_kernel(const float* __restrict__ W, _Float16* __restrict__ Wp,
                              int K, int N, int Kpad, int Npad) {
  const int64_t t = (int64_t)blockIdx.x * blockDim.x + threadIdx.x;
  const int64_t total = (int64_t)Kpad * Npad;
  if (t >= total) return;
  const int k32 = (int)(t & 31);
  const int64_t rest = t >> 5;
  const int n = (int)(rest % Npad);
  const int kb = (int)(rest / Npad);
  const int k = kb * 32 + k32;
  const float v = (k < K && n < N) ? W[(int64_t)k * N + n] : 0.f;
  Wp[t] = (_Float16)v;
}

// --------------------------------------------------------------------------
// WMMA GEMM: C[M,N] f32 = A[M,Kpad] f16 (zero-padded) * Wp (packed f16) + bias.
// One wave per 16x64 output tile; 4 f32 accumulators; branch-free inner loop:
// 2 b128 A-loads + 8 b128 B-loads + 4 v_wmma per 32-wide K step.
// Requires M % 16 == 0 (true for all call sites).
// --------------------------------------------------------------------------
__global__ void __launch_bounds__(32)
wmma_gemm(const _Float16* __restrict__ A, const _Float16* __restrict__ Wp,
          const float* __restrict__ bias, float* __restrict__ C,
          int64_t M, int Kpad, int N, int Npad) {
  const int lane = threadIdx.x;
  const int r = lane & 15;
  const int h = lane >> 4;
  const int h8 = h * 8;
  const int64_t row0 = (int64_t)blockIdx.x * 16;
  const int n0 = blockIdx.y * 64;

  v8f acc[4] = {};
  const _Float16* ap = A + (row0 + r) * (int64_t)Kpad;

  for (int kb = 0; kb < Kpad; kb += 32) {
    const v8h alo = *(const v8h*)(ap + kb + h * 16);
    const v8h ahi = *(const v8h*)(ap + kb + h * 16 + 8);
    const v16h a = __builtin_shufflevector(alo, ahi,
        0, 1, 2, 3, 4, 5, 6, 7, 8, 9, 10, 11, 12, 13, 14, 15);
    const _Float16* wb =
        Wp + (((int64_t)(kb >> 5) * Npad) + n0 + r) * 32 + h * 16;
#pragma unroll
    for (int t = 0; t < 4; ++t) {
      const v8h b0 = *(const v8h*)(wb + t * 16 * 32);
      const v8h b1 = *(const v8h*)(wb + t * 16 * 32 + 8);
      const v16h bf = __builtin_shufflevector(b0, b1,
          0, 1, 2, 3, 4, 5, 6, 7, 8, 9, 10, 11, 12, 13, 14, 15);
      acc[t] = __builtin_amdgcn_wmma_f32_16x16x32_f16(
          false, a, false, bf, (short)0, acc[t], false, false);
    }
  }

#pragma unroll
  for (int t = 0; t < 4; ++t) {
    const int n = n0 + t * 16 + r;
    if (n < N) {
      const float bv = bias[n];
#pragma unroll
      for (int v = 0; v < 8; ++v)
        C[(row0 + v + 8 * h) * (int64_t)N + n] = acc[t][v] + bv;
    }
  }
}

// --------------------------------------------------------------------------
// Farthest point sampling: one workgroup per batch; npoint sequential argmax
// reductions over N points via fixed-order LDS tree (first-max-index).
// --------------------------------------------------------------------------
__global__ void __launch_bounds__(256)
fps_kernel(const float* __restrict__ xyz, int* __restrict__ out_idx,
           float* __restrict__ dist, int Nn, int npoint) {
  const int b = blockIdx.x;
  const int tid = threadIdx.x;
  const float* p = xyz + (int64_t)b * Nn * 3;
  float* d = dist + (int64_t)b * Nn;
  for (int i = tid; i < Nn; i += 256) d[i] = 1e10f;
  __syncthreads();

  __shared__ float sval[256];
  __shared__ int   sidx[256];
  int far = 0;
  for (int it = 0; it < npoint; ++it) {
    if (tid == 0) out_idx[b * npoint + it] = far;
    const float cx = p[far * 3 + 0], cy = p[far * 3 + 1], cz = p[far * 3 + 2];
    float bestv = -1.0f; int besti = 0;
    for (int i = tid; i < Nn; i += 256) {
      const float dx = p[i * 3 + 0] - cx;
      const float dy = p[i * 3 + 1] - cy;
      const float dz = p[i * 3 + 2] - cz;
      const float dd = dx * dx + dy * dy + dz * dz;
      const float nd = fminf(d[i], dd);
      d[i] = nd;
      if (nd > bestv) { bestv = nd; besti = i; }
    }
    sval[tid] = bestv; sidx[tid] = besti;
    __syncthreads();
    for (int s = 128; s > 0; s >>= 1) {
      if (tid < s) {
        if (sval[tid + s] > sval[tid] ||
            (sval[tid + s] == sval[tid] && sidx[tid + s] < sidx[tid])) {
          sval[tid] = sval[tid + s];
          sidx[tid] = sidx[tid + s];
        }
      }
      __syncthreads();
    }
    far = sidx[0];
    __syncthreads();
  }
}

// Gather xyz rows by index: out[t] = xyz[b, idx[t], :]
__global__ void gather3_kernel(const float* __restrict__ xyz, const int* __restrict__ idx,
                               float* __restrict__ out, int Nn, int S, int total) {
  const int t = blockIdx.x * blockDim.x + threadIdx.x;
  if (t >= total) return;
  const int b = t / S;
  const int j = idx[t];
  const float* p = xyz + ((int64_t)b * Nn + j) * 3;
  out[(int64_t)t * 3 + 0] = p[0];
  out[(int64_t)t * 3 + 1] = p[1];
  out[(int64_t)t * 3 + 2] = p[2];
}

// Convert f32 -> f16 elementwise.
__global__ void f32_to_f16_kernel(const float* __restrict__ X, _Float16* __restrict__ Y,
                                  int64_t total) {
  const int64_t t = (int64_t)blockIdx.x * blockDim.x + threadIdx.x;
  if (t >= total) return;
  Y[t] = (_Float16)X[t];
}

// --------------------------------------------------------------------------
// Ball query + grouping -> zero-padded f16 rows of width Cpad:
// [centered xyz (3) | feats (Cf) | zeros]. One block per center; thread 0
// selects first `nsample` in-radius indices (pad with first hit).
// --------------------------------------------------------------------------
__global__ void group_kernel(const float* __restrict__ xyz, const _Float16* __restrict__ feats,
                             const float* __restrict__ centers, _Float16* __restrict__ out,
                             int Nn, int S, int nsample, int Cf, int Cpad, float r2) {
  __shared__ int sel[64];
  const int bs = blockIdx.x;          // b*S + s
  const int b  = bs / S;
  const float cx = centers[(int64_t)bs * 3 + 0];
  const float cy = centers[(int64_t)bs * 3 + 1];
  const float cz = centers[(int64_t)bs * 3 + 2];
  const float* p = xyz + (int64_t)b * Nn * 3;

  if (threadIdx.x == 0) {
    int cnt = 0, first = 0;
    bool have = false;
    for (int j = 0; j < Nn && cnt < nsample; ++j) {
      const float dx = p[j * 3 + 0] - cx;
      const float dy = p[j * 3 + 1] - cy;
      const float dz = p[j * 3 + 2] - cz;
      if (dx * dx + dy * dy + dz * dz <= r2) {
        if (!have) { first = j; have = true; }
        sel[cnt++] = j;
      }
    }
    for (; cnt < nsample; ++cnt) sel[cnt] = first;
  }
  __syncthreads();

  const int k = threadIdx.x;
  const int j = sel[k];
  _Float16* o = out + ((int64_t)bs * nsample + k) * Cpad;
  o[0] = (_Float16)(p[j * 3 + 0] - cx);
  o[1] = (_Float16)(p[j * 3 + 1] - cy);
  o[2] = (_Float16)(p[j * 3 + 2] - cz);
  const _Float16* f = feats + ((int64_t)b * Nn + j) * Cf;
  for (int c = 0; c < Cf; ++c) o[3 + c] = f[c];
  for (int c = 3 + Cf; c < Cpad; ++c) o[c] = (_Float16)0.f;
}

// BN statistics: one block per channel, deterministic LDS reduction.
__global__ void __launch_bounds__(256)
bn_stats_kernel(const float* __restrict__ X, int64_t M, int C,
                float* __restrict__ mean, float* __restrict__ var) {
  const int c = blockIdx.x;
  const int tid = threadIdx.x;
  float s = 0.f, s2 = 0.f;
  for (int64_t i = tid; i < M; i += 256) {
    const float x = X[i * C + c];
    s += x; s2 += x * x;
  }
  __shared__ float sh[256], sh2[256];
  sh[tid] = s; sh2[tid] = s2;
  __syncthreads();
  for (int st = 128; st > 0; st >>= 1) {
    if (tid < st) { sh[tid] += sh[tid + st]; sh2[tid] += sh2[tid + st]; }
    __syncthreads();
  }
  if (tid == 0) {
    const float m = sh[0] / (float)M;
    mean[c] = m;
    var[c] = fmaxf(sh2[0] / (float)M - m * m, 0.f);
  }
}

// Fused BN + ReLU + downconvert: f32 GEMM output -> f16 next-layer input.
__global__ void bn_relu_cast_kernel(const float* __restrict__ X, _Float16* __restrict__ Y,
                                    const float* __restrict__ mean, const float* __restrict__ var,
                                    const float* __restrict__ g, const float* __restrict__ be,
                                    int64_t total, int C) {
  const int64_t t = (int64_t)blockIdx.x * blockDim.x + threadIdx.x;
  if (t >= total) return;
  const int c = (int)(t % C);
  float x = X[t];
  x = g[c] * (x - mean[c]) * rsqrtf(var[c] + BN_EPS) + be[c];
  Y[t] = (_Float16)fmaxf(x, 0.f);
}

// Max over the K (sample) axis: X[BS,Ksz,C] f16 -> Y[BS,C] f16
__global__ void maxpool_kernel(const _Float16* __restrict__ X, _Float16* __restrict__ Y,
                               int64_t BS, int Ksz, int C) {
  const int64_t t = (int64_t)blockIdx.x * blockDim.x + threadIdx.x;
  if (t >= BS * (int64_t)C) return;
  const int64_t bs = t / C;
  const int c = (int)(t % C);
  const _Float16* p = X + bs * (int64_t)Ksz * C + c;
  float m = -3.4e38f;
  for (int k = 0; k < Ksz; ++k) m = fmaxf(m, (float)p[(int64_t)k * C]);
  Y[t] = (_Float16)m;
}

// SA3 concat: [l2_xyz(3) f32 | l2_pts(Cf) f16 | zero pad] -> f16 rows of Cpad.
__global__ void concat_kernel(const float* __restrict__ a3, const _Float16* __restrict__ bC,
                              _Float16* __restrict__ out, int64_t rows, int Cf, int Cpad) {
  const int64_t t = (int64_t)blockIdx.x * blockDim.x + threadIdx.x;
  if (t >= rows * (int64_t)Cpad) return;
  const int64_t r = t / Cpad;
  const int c = (int)(t % Cpad);
  _Float16 v = (_Float16)0.f;
  if (c < 3)            v = (_Float16)a3[r * 3 + c];
  else if (c < 3 + Cf)  v = bC[r * (int64_t)Cf + (c - 3)];
  out[t] = v;
}

// log_softmax per row (B=16 rows of 40), f32 in/out.
__global__ void logsoftmax_kernel(const float* __restrict__ X, float* __restrict__ out,
                                  int Bn, int Cc) {
  const int b = threadIdx.x;
  if (b >= Bn) return;
  const float* x = X + (int64_t)b * Cc;
  float mx = -3.4e38f;
  for (int c = 0; c < Cc; ++c) mx = fmaxf(mx, x[c]);
  float s = 0.f;
  for (int c = 0; c < Cc; ++c) s += expf(x[c] - mx);
  const float lse = mx + logf(s);
  for (int c = 0; c < Cc; ++c) out[(int64_t)b * Cc + c] = x[c] - lse;
}

// --------------------------------------------------------------------------
extern "C" void kernel_launch(void* const* d_in, const int* in_sizes, int n_in,
                              void* d_out, int out_size, void* d_ws, size_t ws_size,
                              hipStream_t stream) {
  (void)in_sizes; (void)n_in; (void)out_size; (void)ws_size;
  const float* xyz = (const float*)d_in[0];
  const float* nrm = (const float*)d_in[1];
  const int SA1 = 2, SA2 = 14, SA3 = 26, FCB = 38;

  // Workspace bump allocator (256B aligned).
  char* w = (char*)d_ws;
  size_t off = 0;
  auto alloc = [&](size_t bytes) -> void* {
    void* p = w + off;
    off += (bytes + 255) & ~(size_t)255;
    return p;
  };
  float*    cbuf  = (float*)   alloc((size_t)33554432 * 4);  // f32 GEMM out (134MB)
  _Float16* a0    = (_Float16*)alloc((size_t)33554432 * 2);  // f16 ping (67MB)
  _Float16* a1    = (_Float16*)alloc((size_t)33554432 * 2);  // f16 pong (67MB)
  _Float16* wpack = (_Float16*)alloc((size_t)1024 * 1024 * 2); // packed weights (2MB)
  float*    dist  = (float*)   alloc((size_t)16 * 4096 * 4);
  int*      fps1  = (int*)     alloc((size_t)16 * 512 * 4);
  float*    l1xyz = (float*)   alloc((size_t)16 * 512 * 3 * 4);
  _Float16* l1pts = (_Float16*)alloc((size_t)16 * 512 * 128 * 2);
  int*      fps2  = (int*)     alloc((size_t)16 * 128 * 4);
  float*    l2xyz = (float*)   alloc((size_t)16 * 128 * 3 * 4);
  _Float16* l2pts = (_Float16*)alloc((size_t)16 * 128 * 256 * 2);
  _Float16* nrm16 = (_Float16*)alloc((size_t)16 * 4096 * 3 * 2);
  float*    mean  = (float*)   alloc((size_t)1024 * 4);
  float*    var   = (float*)   alloc((size_t)1024 * 4);
  _Float16* feat  = (_Float16*)alloc((size_t)16 * 1024 * 2);
  _Float16* fc1h  = (_Float16*)alloc((size_t)16 * 512 * 2);
  _Float16* fc2h  = (_Float16*)alloc((size_t)16 * 256 * 2);

  // One layer: pack weights, WMMA GEMM (f16 in / f32 out), optional BN+ReLU+cast.
  auto run_layer = [&](const float* Wf, const float* bf, const float* gf, const float* bef,
                       const _Float16* Ain, _Float16* Aout, int64_t M, int K, int N,
                       bool do_bn) {
    const int Kpad = (K + 31) & ~31;
    const int Npad = (N + 63) & ~63;
    const int64_t wtot = (int64_t)Kpad * Npad;
    pack_w_kernel<<<(unsigned)((wtot + 255) / 256), 256, 0, stream>>>(Wf, wpack, K, N, Kpad, Npad);
    dim3 grid((unsigned)(M / 16), (unsigned)(Npad / 64));
    wmma_gemm<<<grid, 32, 0, stream>>>(Ain, wpack, bf, cbuf, M, Kpad, N, Npad);
    if (do_bn) {
      bn_stats_kernel<<<N, 256, 0, stream>>>(cbuf, M, N, mean, var);
      const int64_t total = M * N;
      bn_relu_cast_kernel<<<(unsigned)((total + 255) / 256), 256, 0, stream>>>(
          cbuf, Aout, mean, var, gf, bef, total, N);
    }
  };
  auto run_mlp = [&](int pbase, const int* cin, const int* cout, int64_t M,
                     _Float16*& cur, _Float16*& oth) {
    for (int l = 0; l < 3; ++l) {
      run_layer((const float*)d_in[pbase + 4 * l + 0], (const float*)d_in[pbase + 4 * l + 1],
                (const float*)d_in[pbase + 4 * l + 2], (const float*)d_in[pbase + 4 * l + 3],
                cur, oth, M, cin[l], cout[l], true);
      _Float16* t = cur; cur = oth; oth = t;
    }
  };

  // ---------------- SA1: N=4096 -> S=512, nsample=32, r=0.2 ----------------
  fps_kernel<<<16, 256, 0, stream>>>(xyz, fps1, dist, 4096, 512);
  gather3_kernel<<<(16 * 512 + 255) / 256, 256, 0, stream>>>(xyz, fps1, l1xyz, 4096, 512, 16 * 512);
  f32_to_f16_kernel<<<(16 * 4096 * 3 + 255) / 256, 256, 0, stream>>>(nrm, nrm16, (int64_t)16 * 4096 * 3);
  group_kernel<<<16 * 512, 32, 0, stream>>>(xyz, nrm16, l1xyz, a0, 4096, 512, 32, 3, 32, 0.04f);
  {
    _Float16* cur = a0; _Float16* oth = a1;
    const int ci[3] = {6, 64, 64}, co[3] = {64, 64, 128};
    run_mlp(SA1, ci, co, (int64_t)16 * 512 * 32, cur, oth);
    maxpool_kernel<<<(16 * 512 * 128 + 255) / 256, 256, 0, stream>>>(
        cur, l1pts, (int64_t)16 * 512, 32, 128);
  }

  // ---------------- SA2: N=512 -> S=128, nsample=64, r=0.4 -----------------
  fps_kernel<<<16, 256, 0, stream>>>(l1xyz, fps2, dist, 512, 128);
  gather3_kernel<<<(16 * 128 + 255) / 256, 256, 0, stream>>>(l1xyz, fps2, l2xyz, 512, 128, 16 * 128);
  group_kernel<<<16 * 128, 64, 0, stream>>>(l1xyz, l1pts, l2xyz, a0, 512, 128, 64, 128, 160, 0.16f);
  {
    _Float16* cur = a0; _Float16* oth = a1;
    const int ci[3] = {131, 128, 128}, co[3] = {128, 128, 256};
    run_mlp(SA2, ci, co, (int64_t)16 * 128 * 64, cur, oth);
    maxpool_kernel<<<(16 * 128 * 256 + 255) / 256, 256, 0, stream>>>(
        cur, l2pts, (int64_t)16 * 128, 64, 256);
  }

  // ---------------- SA3: group-all over 128 points, 259 channels ----------
  concat_kernel<<<(16 * 128 * 288 + 255) / 256, 256, 0, stream>>>(
      l2xyz, l2pts, a0, (int64_t)16 * 128, 256, 288);
  {
    _Float16* cur = a0; _Float16* oth = a1;
    const int ci[3] = {259, 256, 512}, co[3] = {256, 512, 1024};
    run_mlp(SA3, ci, co, (int64_t)16 * 128, cur, oth);
    maxpool_kernel<<<(16 * 1024 + 255) / 256, 256, 0, stream>>>(
        cur, feat, (int64_t)16, 128, 1024);
  }

  // ---------------- FC head ----------------
  run_layer((const float*)d_in[FCB + 0], (const float*)d_in[FCB + 1],
            (const float*)d_in[FCB + 2], (const float*)d_in[FCB + 3],
            feat, fc1h, 16, 1024, 512, true);
  run_layer((const float*)d_in[FCB + 4], (const float*)d_in[FCB + 5],
            (const float*)d_in[FCB + 6], (const float*)d_in[FCB + 7],
            fc1h, fc2h, 16, 512, 256, true);
  run_layer((const float*)d_in[FCB + 8], (const float*)d_in[FCB + 9],
            nullptr, nullptr, fc2h, nullptr, 16, 256, 40, false);
  logsoftmax_kernel<<<1, 32, 0, stream>>>(cbuf, (float*)d_out, 16, 40);
}
